// DualOrganism_71433896067266
// MI455X (gfx1250) — matrix operand order repaired and checked by reading, not compiled, not run on hardware
//
#include <hip/hip_runtime.h>
#include <hip/hip_bf16.h>
#include <math.h>

#define GRID_ 256
#define NAG   8192
#define DIM   32
#define HID   64
#define KM    15
#define SIGMA 1.5f
#define CELL_CAP 8
#define TW1K  40   // K=33 padded to 40 so transposed rows stay 16B-aligned

typedef __attribute__((ext_vector_type(16))) __bf16 v16bf;
typedef __attribute__((ext_vector_type(8)))  __bf16 v8bf;
typedef __attribute__((ext_vector_type(8)))  float  v8f;

// ---------------- WMMA fragment helpers (CDNA5 wave32 layouts) ----------------
// A-matrix 16-bit 16xK: lane l%16 = row M; lane half hi splits K as
// {hi*8+0..7, 16+hi*8+0..7} -> each fragment = two contiguous 8xbf16 runs.
__device__ __forceinline__ v16bf cat8(v8bf lo, v8bf hh) {
    return __builtin_shufflevector(lo, hh, 0, 1, 2, 3, 4, 5, 6, 7,
                                   8, 9, 10, 11, 12, 13, 14, 15);
}
__device__ __forceinline__ v16bf make_a_bf(const __bf16* m, int ld, int l, int hi, int kb) {
    const __bf16* p = m + l * ld + kb + hi * 8;
    return cat8(*(const v8bf*)p, *(const v8bf*)(p + 16));
}
__device__ __forceinline__ v16bf make_a_f32(const float* m, int ld, int l, int hi, int kb) {
    const float* p = m + l * ld + kb + hi * 8;
    float4 f0 = *(const float4*)(p);
    float4 f1 = *(const float4*)(p + 4);
    float4 f2 = *(const float4*)(p + 16);
    float4 f3 = *(const float4*)(p + 20);
    v16bf a;
    a[0] = (__bf16)f0.x; a[1] = (__bf16)f0.y; a[2] = (__bf16)f0.z; a[3] = (__bf16)f0.w;
    a[4] = (__bf16)f1.x; a[5] = (__bf16)f1.y; a[6] = (__bf16)f1.z; a[7] = (__bf16)f1.w;
    a[8] = (__bf16)f2.x; a[9] = (__bf16)f2.y; a[10] = (__bf16)f2.z; a[11] = (__bf16)f2.w;
    a[12] = (__bf16)f3.x; a[13] = (__bf16)f3.y; a[14] = (__bf16)f3.z; a[15] = (__bf16)f3.w;
    return a;
}
// B-matrix 16-bit KxN: lane l%16 = column N; lane half hi holds K=kb+{hi*8+0..7,
// 16+hi*8+0..7}. Weights staged TRANSPOSED (N-major, K-stride Ks)
// in LDS so each fragment is two contiguous b128 LDS loads.
__device__ __forceinline__ v16bf make_b_t(const __bf16* mT, int Ks, int l, int hi, int kb, int nb) {
    const __bf16* p = mT + (nb + l) * Ks + kb + hi * 8;
    return cat8(*(const v8bf*)p, *(const v8bf*)(p + 16));
}
__device__ __forceinline__ v8f wmma_bf(v16bf a, v16bf b, v8f c) {
    return __builtin_amdgcn_wmma_f32_16x16x32_bf16(false, a, false, b, (short)0, c, false, false);
}
__device__ __forceinline__ v8f zero8() {
    v8f c = {0.f, 0.f, 0.f, 0.f, 0.f, 0.f, 0.f, 0.f};
    return c;
}
// Branch-free tanh: native v_exp_f32 + v_rcp_f32; saturates correctly at +-inf.
__device__ __forceinline__ float fast_tanh(float x) {
    float e = __expf(2.f * x);
    return 1.f - 2.f * __builtin_amdgcn_rcpf(e + 1.f);
}

// ---------------- Gaussian weights (computed per-thread, trivial) ----------------
__device__ __forceinline__ void gauss_w(float* w) {
    float s = 0.f;
#pragma unroll
    for (int t = 0; t < KM; ++t) {
        float ax = (float)t - (KM - 1) * 0.5f;
        w[t] = __expf(-(ax * ax) / (2.f * SIGMA * SIGMA));
        s += w[t];
    }
    float inv = 1.f / s;
#pragma unroll
    for (int t = 0; t < KM; ++t) w[t] *= inv;
}

// ---------------- Kernel 1: zero grid / counters ----------------
__global__ void k_zero(float* egrid, int* cellCount, int* fiCount) {
    int i = blockIdx.x * blockDim.x + threadIdx.x;
    for (int t = i; t < GRID_ * GRID_; t += gridDim.x * blockDim.x) {
        egrid[t] = 0.f;
        cellCount[t] = 0;
    }
    if (i < 2) fiCount[i] = 0;
}

// ---------------- Kernel 2: scatter energies + build cell lists + fi lists ----------------
__global__ void k_scatter(const int* __restrict__ positions, const int* __restrict__ roles,
                          const int* __restrict__ org_ids, const float* __restrict__ energies,
                          float* egrid, int* cellCount, int* cellList, int* fiCount, int* fiList) {
    int i = blockIdx.x * blockDim.x + threadIdx.x;
    if (i >= NAG) return;
    int x = positions[2 * i], y = positions[2 * i + 1];
    int c = y * GRID_ + x;
    egrid[c] = energies[i];
    int slot = atomicAdd(&cellCount[c], 1);
    if (slot < CELL_CAP) cellList[c * CELL_CAP + slot] = i;
    if (roles[i] == 1) {
        int o = org_ids[i];
        int s = atomicAdd(&fiCount[o], 1);
        fiList[o * NAG + s] = i;
    }
}

// ---------------- Kernels 3/4: separable 15-tap Gaussian ('SAME', zero pad) ----------------
__global__ void k_blur_h(const float* __restrict__ src, float* __restrict__ dst) {
    int idx = blockIdx.x * blockDim.x + threadIdx.x;
    if (idx >= GRID_ * GRID_) return;
    int y = idx >> 8, x = idx & 255;
    float w[KM];
    gauss_w(w);
    float s = 0.f;
#pragma unroll
    for (int t = 0; t < KM; ++t) {
        int xx = x + t - 7;
        if (xx >= 0 && xx < GRID_) s += w[t] * src[y * GRID_ + xx];
    }
    dst[idx] = s;
}
__global__ void k_blur_v(const float* __restrict__ src, float* __restrict__ dst) {
    int idx = blockIdx.x * blockDim.x + threadIdx.x;
    if (idx >= GRID_ * GRID_) return;
    int y = idx >> 8, x = idx & 255;
    float w[KM];
    gauss_w(w);
    float s = 0.f;
#pragma unroll
    for (int t = 0; t < KM; ++t) {
        int yy = y + t - 7;
        if (yy >= 0 && yy < GRID_) s += w[t] * src[yy * GRID_ + x];
    }
    dst[idx] = s;
}

// ---------------- Kernel 5: gather potential per agent ----------------
__global__ void k_potential(const int* __restrict__ positions, const float* __restrict__ field,
                            float* __restrict__ potential) {
    int i = blockIdx.x * blockDim.x + threadIdx.x;
    if (i >= NAG) return;
    potential[i] = field[positions[2 * i + 1] * GRID_ + positions[2 * i]];
}

// ---------------- Kernel 6: p_out / p_in = tanh(states @ w[org]) via WMMA ----------------
__global__ void __launch_bounds__(256) k_proj(const float* __restrict__ states,
                                              const int* __restrict__ org_ids,
                                              const float* __restrict__ w_out,
                                              const float* __restrict__ w_in,
                                              float* __restrict__ p_out,
                                              float* __restrict__ p_in) {
    __shared__ __bf16 lwT[2][2][HID * DIM];  // [mat][org], transposed N-major, 16 KB
    for (int t = threadIdx.x; t < 2 * 2 * DIM * HID; t += 256) {
        int mat = t >> 12;
        int rem = t & 4095;
        int o = rem >> 11;
        int k = (rem & 2047) >> 6;
        int n = rem & 63;
        float v = mat ? w_in[rem] : w_out[rem];
        lwT[mat][o][n * DIM + k] = (__bf16)v;
    }
    __syncthreads();

    int wave = threadIdx.x >> 5, lane = threadIdx.x & 31;
    int l = lane & 15, hi = lane >> 4;
    int tile = blockIdx.x * 8 + wave;  // 512 tiles total

    const float* arow = states + (size_t)tile * 16 * DIM;
    v16bf A = make_a_f32(arow, DIM, l, hi, 0);  // full K=32 in one fragment

    int orr[8];
#pragma unroll
    for (int r = 0; r < 8; ++r) orr[r] = org_ids[tile * 16 + r + 8 * hi];

#pragma unroll
    for (int mat = 0; mat < 2; ++mat) {
        float* dst = mat ? p_in : p_out;
#pragma unroll
        for (int nt = 0; nt < 4; ++nt) {
            v8f c0 = wmma_bf(A, make_b_t(&lwT[mat][0][0], DIM, l, hi, 0, nt * 16), zero8());
            v8f c1 = wmma_bf(A, make_b_t(&lwT[mat][1][0], DIM, l, hi, 0, nt * 16), zero8());
#pragma unroll
            for (int r = 0; r < 8; ++r) {
                int row = tile * 16 + r + 8 * hi;
                float v = (orr[r] == 0) ? c0[r] : c1[r];
                dst[(size_t)row * HID + nt * 16 + l] = fast_tanh(v);
            }
        }
    }
}

// ---------------- Kernel 7: self-pattern + transform MLP chain via WMMA ----------------
__global__ void __launch_bounds__(256) k_mlp(const float* __restrict__ states,
                                             const int* __restrict__ org_ids,
                                             const float* __restrict__ sw1, const float* __restrict__ sb1,
                                             const float* __restrict__ sw2, const float* __restrict__ sb2,
                                             const float* __restrict__ tw1, const float* __restrict__ tb1,
                                             const float* __restrict__ tw2, const float* __restrict__ tb2,
                                             const float* __restrict__ potential,
                                             float* __restrict__ transformed) {
    // all weights staged transposed (N-major) as bf16
    __shared__ __bf16 Lsw1T[2 * HID * DIM];    // 8 KB   [o][n<64][k<32]
    __shared__ __bf16 Lsw2T[2 * DIM * HID];    // 8 KB   [o][n<32][k<64]
    __shared__ __bf16 Ltw1T[2 * HID * TW1K];   // 10 KB  [o][n<64][k<40] (k=32 = potential row)
    __shared__ __bf16 Ltw2T[2 * DIM * HID];    // 8 KB   [o][n<32][k<64]
    __shared__ float Lsb1[2 * HID], Lsb2[2 * DIM], Ltb1[2 * HID], Ltb2[2 * DIM];
    __shared__ __bf16 scr[8][16 * HID];        // wave-private scratch, 16 KB

    for (int t = threadIdx.x; t < 2 * DIM * HID; t += 256) {
        int o = t >> 11, k = (t & 2047) >> 6, n = t & 63;
        Lsw1T[(o * HID + n) * DIM + k] = (__bf16)sw1[t];
    }
    for (int t = threadIdx.x; t < 2 * HID * DIM; t += 256) {
        int o = t >> 11, k = (t & 2047) >> 5, n = t & 31;
        Lsw2T[(o * DIM + n) * HID + k] = (__bf16)sw2[t];
        int o2 = o, k2 = k, n2 = n;
        Ltw2T[(o2 * DIM + n2) * HID + k2] = (__bf16)tw2[t];
    }
    for (int t = threadIdx.x; t < 2 * (DIM + 1) * HID; t += 256) {
        int o = t / ((DIM + 1) * HID);
        int r = t - o * (DIM + 1) * HID;
        int k = r >> 6, n = r & 63;
        Ltw1T[(o * HID + n) * TW1K + k] = (__bf16)tw1[t];
    }
    for (int t = threadIdx.x; t < 2 * HID; t += 256) { Lsb1[t] = sb1[t]; Ltb1[t] = tb1[t]; }
    for (int t = threadIdx.x; t < 2 * DIM; t += 256) { Lsb2[t] = sb2[t]; Ltb2[t] = tb2[t]; }
    __syncthreads();

    int wave = threadIdx.x >> 5, lane = threadIdx.x & 31;
    int l = lane & 15, hi = lane >> 4;
    int tile = blockIdx.x * 8 + wave;
    __bf16* sc = scr[wave];

    const float* arow = states + (size_t)tile * 16 * DIM;
    v16bf Ast = make_a_f32(arow, DIM, l, hi, 0);

    int orr[8];
    float pot[8];
#pragma unroll
    for (int r = 0; r < 8; ++r) {
        int row = tile * 16 + r + 8 * hi;
        orr[r] = org_ids[row];
        pot[r] = potential[row];
    }

    for (int o = 0; o < 2; ++o) {
        // h1 = tanh(states @ sw1[o] + sb1[o])  (16x64) -> LDS scratch
#pragma unroll
        for (int nt = 0; nt < 4; ++nt) {
            v8f c = wmma_bf(Ast, make_b_t(Lsw1T + o * HID * DIM, DIM, l, hi, 0, nt * 16), zero8());
#pragma unroll
            for (int r = 0; r < 8; ++r)
                sc[(r + 8 * hi) * HID + nt * 16 + l] =
                    (__bf16)fast_tanh(c[r] + Lsb1[o * HID + nt * 16 + l]);
        }
        // self_pat = h1 @ sw2[o] + sb2[o]; enriched = states + 0.1*self_pat  (16x32)
        v16bf Ah0 = make_a_bf(sc, HID, l, hi, 0);
        v16bf Ah1 = make_a_bf(sc, HID, l, hi, 32);
        float enr[2][8];
#pragma unroll
        for (int nt = 0; nt < 2; ++nt) {
            v8f c = wmma_bf(Ah0, make_b_t(Lsw2T + o * DIM * HID, HID, l, hi, 0, nt * 16), zero8());
            c = wmma_bf(Ah1, make_b_t(Lsw2T + o * DIM * HID, HID, l, hi, 32, nt * 16), c);
#pragma unroll
            for (int r = 0; r < 8; ++r) {
                int row = tile * 16 + r + 8 * hi;
                float sp = c[r] + Lsb2[o * DIM + nt * 16 + l];
                enr[nt][r] = states[(size_t)row * DIM + nt * 16 + l] + 0.1f * sp;
            }
        }
#pragma unroll
        for (int nt = 0; nt < 2; ++nt)
#pragma unroll
            for (int r = 0; r < 8; ++r)
                sc[(r + 8 * hi) * DIM + nt * 16 + l] = (__bf16)enr[nt][r];
        v16bf Aen = make_a_bf(sc, DIM, l, hi, 0);
        // h3 = tanh(enriched @ tw1[o][0:32] + potential * tw1[o][32] + tb1[o])  (16x64)
        // 33rd input column handled as rank-1 update (no K padding needed)
#pragma unroll
        for (int nt = 0; nt < 4; ++nt) {
            v8f c = wmma_bf(Aen, make_b_t(Ltw1T + o * HID * TW1K, TW1K, l, hi, 0, nt * 16), zero8());
#pragma unroll
            for (int r = 0; r < 8; ++r) {
                float w32 = (float)Ltw1T[(o * HID + nt * 16 + l) * TW1K + DIM];
                float v = c[r] + pot[r] * w32 + Ltb1[o * HID + nt * 16 + l];
                sc[(r + 8 * hi) * HID + nt * 16 + l] = (__bf16)fast_tanh(v);
            }
        }
        v16bf Ah3a = make_a_bf(sc, HID, l, hi, 0);
        v16bf Ah3b = make_a_bf(sc, HID, l, hi, 32);
        // transformed = h3 @ tw2[o] + tb2[o]  (16x32); write rows matching org o
#pragma unroll
        for (int nt = 0; nt < 2; ++nt) {
            v8f c = wmma_bf(Ah3a, make_b_t(Ltw2T + o * DIM * HID, HID, l, hi, 0, nt * 16), zero8());
            c = wmma_bf(Ah3b, make_b_t(Ltw2T + o * DIM * HID, HID, l, hi, 32, nt * 16), c);
#pragma unroll
            for (int r = 0; r < 8; ++r) {
                int row = tile * 16 + r + 8 * hi;
                if (orr[r] == o)
                    transformed[(size_t)row * DIM + nt * 16 + l] = c[r] + Ltb2[o * DIM + nt * 16 + l];
            }
        }
    }
}

// ---------------- Kernel 8: neighbor stats via spatial hash ----------------
__global__ void k_neighbors(const int* __restrict__ positions, const int* __restrict__ roles,
                            const int* __restrict__ org_ids,
                            const int* __restrict__ cellCount, const int* __restrict__ cellList,
                            const float* __restrict__ p_out, const float* __restrict__ p_in,
                            float* cnt_o, float* smass_o, float* sfi_o, float* ofi_o,
                            float* so_o, float* si_o) {
    int i = blockIdx.x * blockDim.x + threadIdx.x;
    if (i >= NAG) return;
    int xi = positions[2 * i], yi = positions[2 * i + 1];
    int org = org_ids[i];
    float cnt = 0.f, sm = 0.f, sf = 0.f, of = 0.f, so = 0.f, si = 0.f;
    const float4* pio = (const float4*)(p_out + (size_t)i * HID);
    const float4* pii = (const float4*)(p_in + (size_t)i * HID);
    __builtin_prefetch(pio, 0, 3);
    __builtin_prefetch(pii, 0, 3);
    for (int dy = -5; dy <= 5; ++dy) {
        int ny = yi + dy;
        if (ny < 0 || ny >= GRID_) continue;
        for (int dx = -5; dx <= 5; ++dx) {
            if (dx * dx + dy * dy > 25) continue;  // RADIUS2
            int nx = xi + dx;
            if (nx < 0 || nx >= GRID_) continue;
            int c = ny * GRID_ + nx;
            int n = cellCount[c];
            if (n > CELL_CAP) n = CELL_CAP;
            for (int s = 0; s < n; ++s) {
                int j = cellList[c * CELL_CAP + s];
                if (j == i) continue;  // not_self
                int fi = (roles[j] == 1);
                if (org_ids[j] == org) {
                    cnt += 1.f;
                    sm += (roles[j] == 0) ? 1.f : 0.f;
                    sf += (float)fi;
                    const float4* pjo = (const float4*)(p_out + (size_t)j * HID);
                    const float4* pji = (const float4*)(p_in + (size_t)j * HID);
                    float a = 0.f, b = 0.f;
#pragma unroll
                    for (int t = 0; t < HID / 4; ++t) {
                        float4 u = pio[t], v = pjo[t];
                        a += u.x * v.x + u.y * v.y + u.z * v.z + u.w * v.w;
                        float4 p = pii[t], q = pji[t];
                        b += p.x * q.x + p.y * q.y + p.z * q.z + p.w * q.w;
                    }
                    so += a;
                    si += b;
                } else {
                    of += (float)fi;
                }
            }
        }
    }
    cnt_o[i] = cnt; smass_o[i] = sm; sfi_o[i] = sf; ofi_o[i] = of; so_o[i] = so; si_o[i] = si;
}

// ---------------- Kernel 9: global nearest same-org fi agent (wave per agent) ----------------
__global__ void k_nearest(const int* __restrict__ positions, const int* __restrict__ org_ids,
                          const int* __restrict__ fiCount, const int* __restrict__ fiList,
                          int* __restrict__ nearest, int* __restrict__ hasfi) {
    int gw = (blockIdx.x * blockDim.x + threadIdx.x) >> 5;
    int lane = threadIdx.x & 31;
    if (gw >= NAG) return;
    int i = gw;
    int org = org_ids[i];
    int xi = positions[2 * i], yi = positions[2 * i + 1];
    int n = fiCount[org];
    if (n > NAG) n = NAG;
    const int* list = fiList + org * NAG;
    int bd = 0x7fffffff, bj = 0x7fffffff;
    for (int t = lane; t < n; t += 32) {
        int j = list[t];
        int dx = positions[2 * j] - xi, dy = positions[2 * j + 1] - yi;
        int d2 = dx * dx + dy * dy;
        if (d2 < bd || (d2 == bd && j < bj)) { bd = d2; bj = j; }
    }
    // wave32 lexicographic (d2, index) min-reduction: matches argmin tie-breaking
#pragma unroll
    for (int off = 16; off > 0; off >>= 1) {
        int od = __shfl_xor(bd, off, 32);
        int oj = __shfl_xor(bj, off, 32);
        if (od < bd || (od == bd && oj < bj)) { bd = od; bj = oj; }
    }
    if (lane == 0) {
        nearest[i] = (n > 0) ? bj : 0;
        hasfi[i] = (n > 0) ? 1 : 0;
    }
}

// ---------------- Kernel 10: combine everything, write all 5 outputs ----------------
__global__ void k_finalize(const int* __restrict__ positions, const int* __restrict__ roles,
                           const float* __restrict__ energies, const int* __restrict__ org_ids,
                           const float* __restrict__ rand_switch,
                           const float* __restrict__ states, const float* __restrict__ transformed,
                           const float* __restrict__ potential,
                           const float* __restrict__ cnt_a, const float* __restrict__ smass,
                           const float* __restrict__ sfi, const float* __restrict__ ofi,
                           const float* __restrict__ so, const float* __restrict__ si,
                           const int* __restrict__ nearest, const int* __restrict__ hasfi,
                           float* __restrict__ out) {
    int i = blockIdx.x * blockDim.x + threadIdx.x;
    if (i >= NAG) return;
    float cnt = cnt_a[i];
    int has_nb = cnt > 0.f;
    float cs = fmaxf(cnt, 1.f);
    float net_inf = has_nb ? (so[i] - si[i]) / cs : 0.f;

    // new_states (float4-vectorized)
    float4* ns = (float4*)(out + (size_t)i * DIM);
    const float4* st = (const float4*)(states + (size_t)i * DIM);
    const float4* tr = (const float4*)(transformed + (size_t)i * DIM);
#pragma unroll
    for (int t = 0; t < DIM / 4; ++t) {
        float4 s4 = st[t];
        float4 t4 = tr[t];
        float4 o4;
        o4.x = has_nb ? (t4.x + 0.3f * s4.x) : s4.x;
        o4.y = has_nb ? (t4.y + 0.3f * s4.y) : s4.y;
        o4.z = has_nb ? (t4.z + 0.3f * s4.z) : s4.z;
        o4.w = has_nb ? (t4.w + 0.3f * s4.w) : s4.w;
        ns[t] = o4;
    }

    float e = energies[i], pot = potential[i];
    float sm = smass[i], sf = sfi[i], of = ofi[i];
    float fi_e = e + 0.02f * sm - 0.03f * of;
    float mass_e = e * 0.995f + 0.05f * fmaxf(pot, 0.f) +
                   0.02f * (sf > 0.f ? 1.f : 0.f) - 0.01f * (of > 0.f ? 1.f : 0.f);
    float ne = (roles[i] == 1) ? fi_e : mass_e;
    ne += 0.02f * fmaxf(net_inf + 0.5f, 0.f);
    ne = fminf(fmaxf(ne, 0.f), 1.f);
    float score = net_inf + 0.5f;

    int can_fi = (ne > 0.5f) && (sm >= 2.f) && ((sf == 0.f) || (score > 0.3f)) && (of == 0.f);
    int loses = (sm < 1.f) || (ne < 0.2f) || ((of > 0.f) && (of >= sf));
    int role = roles[i];
    int nr = (role == 0) ? (can_fi ? 1 : 0) : ((role == 1) ? (loses ? 0 : 1) : role);

    int hf = hasfi[i];
    int mv = (nr == 0) && hf;
    int x = positions[2 * i], y = positions[2 * i + 1];
    int nj = nearest[i];
    if (nj < 0 || nj >= NAG) nj = 0;
    int fx = positions[2 * nj], fy = positions[2 * nj + 1];
    int dx = (fx > x) - (fx < x), dy = (fy > y) - (fy < y);
    int nx = x + (mv ? dx : 0);
    nx = nx < 0 ? 0 : (nx > GRID_ - 1 ? GRID_ - 1 : nx);
    int nyv = y + (mv ? dy : 0);
    nyv = nyv < 0 ? 0 : (nyv > GRID_ - 1 ? GRID_ - 1 : nyv);

    int defect = (role == 0) && (sf == 0.f) && (of >= 2.f) && (rand_switch[i] < 0.1f);
    int no = defect ? (1 - org_ids[i]) : org_ids[i];

    float* o_e = out + (size_t)NAG * DIM;
    o_e[i] = ne;
    float* o_r = o_e + NAG;
    o_r[i] = (float)nr;
    float* o_p = o_r + NAG;
    o_p[2 * i] = (float)nx;
    o_p[2 * i + 1] = (float)nyv;
    float* o_o = o_p + 2 * NAG;
    o_o[i] = (float)no;
}

// ---------------- host launcher ----------------
extern "C" void kernel_launch(void* const* d_in, const int* in_sizes, int n_in,
                              void* d_out, int out_size, void* d_ws, size_t ws_size,
                              hipStream_t stream) {
    const int* positions = (const int*)d_in[0];
    const float* states = (const float*)d_in[1];
    const int* roles = (const int*)d_in[2];
    const float* energies = (const float*)d_in[3];
    const int* org_ids = (const int*)d_in[4];
    const float* rand_switch = (const float*)d_in[5];
    const float* w_out = (const float*)d_in[6];
    const float* w_in = (const float*)d_in[7];
    const float* sw1 = (const float*)d_in[8];
    const float* sb1 = (const float*)d_in[9];
    const float* sw2 = (const float*)d_in[10];
    const float* sb2 = (const float*)d_in[11];
    const float* tw1 = (const float*)d_in[12];
    const float* tb1 = (const float*)d_in[13];
    const float* tw2 = (const float*)d_in[14];
    const float* tb2 = (const float*)d_in[15];
    (void)in_sizes; (void)n_in; (void)out_size; (void)ws_size;

    char* ws = (char*)d_ws;
    size_t off = 0;
    auto alloc = [&](size_t bytes) -> char* {
        char* p = ws + off;
        off = (off + bytes + 255) & ~(size_t)255;
        return p;
    };
    float* egrid = (float*)alloc(GRID_ * GRID_ * sizeof(float));
    float* tmp = (float*)alloc(GRID_ * GRID_ * sizeof(float));
    float* field = (float*)alloc(GRID_ * GRID_ * sizeof(float));
    int* cellCount = (int*)alloc(GRID_ * GRID_ * sizeof(int));
    int* cellList = (int*)alloc((size_t)GRID_ * GRID_ * CELL_CAP * sizeof(int));
    int* fiCount = (int*)alloc(2 * sizeof(int));
    int* fiList = (int*)alloc(2 * NAG * sizeof(int));
    float* potential = (float*)alloc(NAG * sizeof(float));
    float* p_out = (float*)alloc((size_t)NAG * HID * sizeof(float));
    float* p_in = (float*)alloc((size_t)NAG * HID * sizeof(float));
    float* transformed = (float*)alloc((size_t)NAG * DIM * sizeof(float));
    float* cnt_a = (float*)alloc(NAG * sizeof(float));
    float* smass = (float*)alloc(NAG * sizeof(float));
    float* sfi = (float*)alloc(NAG * sizeof(float));
    float* ofi = (float*)alloc(NAG * sizeof(float));
    float* so = (float*)alloc(NAG * sizeof(float));
    float* si = (float*)alloc(NAG * sizeof(float));
    int* nearest = (int*)alloc(NAG * sizeof(int));
    int* hasfi = (int*)alloc(NAG * sizeof(int));

    k_zero<<<256, 256, 0, stream>>>(egrid, cellCount, fiCount);
    k_scatter<<<NAG / 256, 256, 0, stream>>>(positions, roles, org_ids, energies,
                                             egrid, cellCount, cellList, fiCount, fiList);
    k_blur_h<<<GRID_ * GRID_ / 256, 256, 0, stream>>>(egrid, tmp);
    k_blur_v<<<GRID_ * GRID_ / 256, 256, 0, stream>>>(tmp, field);
    k_potential<<<NAG / 256, 256, 0, stream>>>(positions, field, potential);
    k_proj<<<NAG / 16 / 8, 256, 0, stream>>>(states, org_ids, w_out, w_in, p_out, p_in);
    k_mlp<<<NAG / 16 / 8, 256, 0, stream>>>(states, org_ids, sw1, sb1, sw2, sb2,
                                            tw1, tb1, tw2, tb2, potential, transformed);
    k_neighbors<<<NAG / 256, 256, 0, stream>>>(positions, roles, org_ids, cellCount, cellList,
                                               p_out, p_in, cnt_a, smass, sfi, ofi, so, si);
    k_nearest<<<NAG * 32 / 256, 256, 0, stream>>>(positions, org_ids, fiCount, fiList,
                                                  nearest, hasfi);
    k_finalize<<<NAG / 256, 256, 0, stream>>>(positions, roles, energies, org_ids, rand_switch,
                                              states, transformed, potential,
                                              cnt_a, smass, sfi, ofi, so, si,
                                              nearest, hasfi, (float*)d_out);
}
